// TorchEncoderBlock_4647154614438
// MI455X (gfx1250) — matrix operand order repaired and checked by reading, not compiled, not run on hardware
//
#include <hip/hip_runtime.h>
#include <hip/hip_bf16.h>
#include <math.h>

#define HID 1152
#define MLP_D 4304
#define NH 16
#define HD 72
#define SEQ 3072
#define SEGS 6
#define HDP_Q 96   // padded head dim for QK^T reduction (multiple of 32)
#define HDP_V 80   // padded head dim for P*V output (multiple of 16)

typedef __bf16 bf16_t;
typedef __bf16 v16bf __attribute__((ext_vector_type(16)));
typedef __bf16 v8bf  __attribute__((ext_vector_type(8)));
typedef float  v8f   __attribute__((ext_vector_type(8)));

static __device__ inline bf16_t tobf(float f){ return (bf16_t)f; }

// ---- async global -> LDS copies (CDNA5 GLOBAL_LOAD_ASYNC_TO_LDS_B128, ASYNCcnt) ----
#if defined(__HIP_DEVICE_COMPILE__) && \
    __has_builtin(__builtin_amdgcn_global_load_async_to_lds_b128) && \
    __has_builtin(__builtin_amdgcn_s_wait_asynccnt)
#define HAVE_ASYNC_LDS 1
#else
#define HAVE_ASYNC_LDS 0
#endif

#if HAVE_ASYNC_LDS
typedef int i4v __attribute__((vector_size(4 * sizeof(int))));
typedef __attribute__((address_space(1))) i4v* gbl128_t;
typedef __attribute__((address_space(3))) i4v* lds128_t;
#endif

static __device__ inline void async_cp16(void* lds, const void* g){
#if HAVE_ASYNC_LDS
  __builtin_amdgcn_global_load_async_to_lds_b128(
      (gbl128_t)(unsigned long long)g,
      (lds128_t)(unsigned int)(unsigned long long)lds,
      0, 0);
#else
  *(v8bf*)lds = *(const v8bf*)g;
#endif
}
static __device__ inline void async_wait0(){
#if HAVE_ASYNC_LDS
  __builtin_amdgcn_s_wait_asynccnt(0);
#endif
}

// ---- WMMA fragment loaders, vectorized (2 x ds_load_b128 each) ----
static __device__ inline v16bf load_a_frag(const bf16_t* base, int ld, int lane){
  int m = lane & 15, hi = lane >> 4;
  const v8bf lo = *(const v8bf*)(base + m*ld + hi*8);
  const v8bf hp = *(const v8bf*)(base + m*ld + 16 + hi*8);
  return __builtin_shufflevector(lo, hp, 0,1,2,3,4,5,6,7,8,9,10,11,12,13,14,15);
}
// B stored transposed (N-major): b[j<8] = sBT[n*ld + hi*8 + j], b[j>=8] = sBT[n*ld + 16 + hi*8 + j-8].
static __device__ inline v16bf load_bt_frag(const bf16_t* base, int ld, int lane){
  int n = lane & 15, koff = (lane>>4)*8;
  const v8bf lo = *(const v8bf*)(base + n*ld + koff);
  const v8bf hp = *(const v8bf*)(base + n*ld + 16 + koff);
  return __builtin_shufflevector(lo, hp, 0,1,2,3,4,5,6,7,8,9,10,11,12,13,14,15);
}
static __device__ inline v8f wmma_bf16(v16bf a, v16bf b, v8f c){
  return __builtin_amdgcn_wmma_f32_16x16x32_bf16(false, a, false, b, (short)0, c, false, false);
}

// ---- fp32 [KxN] -> bf16 transposed [NxK], LDS-tiled so both sides coalesce ----
__global__ void cvt_transpose_bf16_kernel(const float* __restrict__ src, bf16_t* __restrict__ dst,
                                          int K, int N){
  __shared__ float tile[32][33];
  int kb = blockIdx.x*32, nb = blockIdx.y*32;
  int tx = threadIdx.x & 31, ty = threadIdx.x >> 5;  // 256 thr: ty 0..7
#pragma unroll
  for(int r=ty; r<32; r+=8){
    int k = kb+r, n = nb+tx;
    tile[r][tx] = (k<K && n<N) ? src[(size_t)k*N+n] : 0.f;
  }
  __syncthreads();
#pragma unroll
  for(int r=ty; r<32; r+=8){
    int n = nb+r, k = kb+tx;
    if(n<N && k<K) dst[(size_t)n*K + k] = tobf(tile[tx][r]);
  }
}

// ---- LayerNorm (fp32 in) -> bf16 out, one block per row ----
__global__ void ln_bf16_kernel(const float* __restrict__ X, const float* __restrict__ g,
                               const float* __restrict__ b, bf16_t* __restrict__ O){
  __shared__ float red[256];
  int row = blockIdx.x, tid = threadIdx.x;
  const float* x = X + (size_t)row*HID;
  float s=0.f;
  for(int c=tid;c<HID;c+=256) s += x[c];
  red[tid]=s; __syncthreads();
  for(int o=128;o>0;o>>=1){ if(tid<o) red[tid]+=red[tid+o]; __syncthreads(); }
  float mu = red[0]*(1.0f/HID);
  __syncthreads();
  float v=0.f;
  for(int c=tid;c<HID;c+=256){ float d=x[c]-mu; v+=d*d; }
  red[tid]=v; __syncthreads();
  for(int o=128;o>0;o>>=1){ if(tid<o) red[tid]+=red[tid+o]; __syncthreads(); }
  float rstd = rsqrtf(red[0]*(1.0f/HID) + 1e-5f);
  for(int c=tid;c<HID;c+=256)
    O[(size_t)row*HID + c] = tobf((x[c]-mu)*rstd*g[c] + b[c]);
}

// ---- bf16 GEMM: C = A[MxK] * BT[NxK]^T + bias. Block 64x128, wave 32x32, async double-buffer.
// MODE 0: f32 ; 1: f32 + residual ; 2: gelu->bf16 ; 3: bf16
template<int MODE>
__global__ void gemm_bf16_kernel(const bf16_t* __restrict__ A, const bf16_t* __restrict__ BT,
                                 const float* __restrict__ bias, const float* __restrict__ R,
                                 float* __restrict__ Cf, bf16_t* __restrict__ Cb,
                                 int M, int N, int K){
  constexpr int BM=64, BN=128, BK=32;
  __shared__ alignas(16) bf16_t sA [2][BM*BK];
  __shared__ alignas(16) bf16_t sBT[2][BN*BK];
  int tid = threadIdx.x, lane = tid & 31, wave = tid >> 5;
  int n0 = blockIdx.x * BN, m0 = blockIdx.y * BM;
  int wm = wave & 1, wn = wave >> 1;
  v8f acc[2][2] = {};
  int ksteps = (K + BK - 1) / BK;

  auto stage = [&](int buf, int ks){
    int k0 = ks*BK;
    { // A tile 64x32 : 256 vec8 chunks, 1/thread
      int r = tid >> 2, c8 = tid & 3;
      int gk = k0 + c8*8;
      if(gk + 8 <= K) async_cp16(&sA[buf][r*BK + c8*8], &A[(size_t)(m0+r)*K + gk]);
      else { v8bf v = {};
        for(int e=0;e<8;++e) if(gk+e<K) v[e] = A[(size_t)(m0+r)*K + gk+e];
        *(v8bf*)&sA[buf][r*BK + c8*8] = v; }
    }
#pragma unroll
    for(int t=0;t<2;++t){ // BT tile 128x32 : 512 vec8 chunks, straight copy
      int chunk = tid + t*256;
      int n = chunk >> 2, c8 = chunk & 3;
      int gn = n0 + n, gk = k0 + c8*8;
      if(gn < N && gk + 8 <= K) async_cp16(&sBT[buf][n*BK + c8*8], &BT[(size_t)gn*K + gk]);
      else { v8bf v = {};
        if(gn < N) for(int e=0;e<8;++e) if(gk+e<K) v[e] = BT[(size_t)gn*K + gk+e];
        *(v8bf*)&sBT[buf][n*BK + c8*8] = v; }
    }
  };

  stage(0, 0); async_wait0(); __syncthreads();
  for(int ks=0; ks<ksteps; ++ks){
    int buf = ks & 1;
    if(ks+1 < ksteps) stage(buf^1, ks+1);   // overlap next tile DMA with compute
    v16bf a0 = load_a_frag (sA [buf] + (wm*32     )*BK, BK, lane);
    v16bf a1 = load_a_frag (sA [buf] + (wm*32 + 16)*BK, BK, lane);
    v16bf b0 = load_bt_frag(sBT[buf] + (wn*32     )*BK, BK, lane);
    v16bf b1 = load_bt_frag(sBT[buf] + (wn*32 + 16)*BK, BK, lane);
    acc[0][0] = wmma_bf16(a0, b0, acc[0][0]);
    acc[0][1] = wmma_bf16(a0, b1, acc[0][1]);
    acc[1][0] = wmma_bf16(a1, b0, acc[1][0]);
    acc[1][1] = wmma_bf16(a1, b1, acc[1][1]);
    async_wait0(); __syncthreads();
  }
  int mh = (lane>>4)*8;
#pragma unroll
  for(int tn=0; tn<2; ++tn){
    int col = n0 + wn*32 + tn*16 + (lane&15);
    if(col < N){
      float bb = bias ? bias[col] : 0.0f;
#pragma unroll
      for(int tm=0; tm<2; ++tm){
#pragma unroll
        for(int i=0;i<8;++i){
          int row = m0 + wm*32 + tm*16 + mh + i;
          float v = acc[tm][tn][i] + bb;
          if(MODE==0)      Cf[(size_t)row*N + col] = v;
          else if(MODE==1) Cf[(size_t)row*N + col] = R[(size_t)row*N + col] + v;
          else if(MODE==2){
            float gl = 0.5f*v*(1.0f+tanhf(0.7978845608f*(v+0.044715f*v*v*v)));
            Cb[(size_t)row*N + col] = tobf(gl);
          } else           Cb[(size_t)row*N + col] = tobf(v);
        }
      }
    }
  }
}

// ---- RoPE + scatter: Q[h][S][96], K[h][S][96] (row-major), V^T[h][80][S] ----
__global__ void rope_scatter_kernel(const float* __restrict__ qkv, const float* __restrict__ fc,
                                    bf16_t* __restrict__ Q, bf16_t* __restrict__ Kb,
                                    bf16_t* __restrict__ VT){
  int t = blockIdx.x*blockDim.x + threadIdx.x;
  int total = SEQ*NH*48;
  if(t>=total) return;
  int j = t % 48;
  int h = (t/48) % NH;
  int s = t/(48*NH);
  int d0 = 2*j;
  size_t qrow = ((size_t)h*SEQ + s)*HDP_Q;
  if(j < HD/2){
    float c  = fc[((size_t)s*(HD/2)+j)*2+0];
    float sn = fc[((size_t)s*(HD/2)+j)*2+1];
    const float* base = qkv + (size_t)s*(3*HID);
    float qa = base[h*HD + d0],        qb = base[h*HD + d0 + 1];
    float ka = base[HID + h*HD + d0],  kb = base[HID + h*HD + d0 + 1];
    Q [qrow + d0]   = tobf(qa*c - qb*sn);
    Q [qrow + d0+1] = tobf(qa*sn + qb*c);
    Kb[qrow + d0]   = tobf(ka*c - kb*sn);
    Kb[qrow + d0+1] = tobf(ka*sn + kb*c);
    float va = base[2*HID + h*HD + d0], vb = base[2*HID + h*HD + d0 + 1];
    VT[((size_t)h*HDP_V + d0  )*SEQ + s] = tobf(va);
    VT[((size_t)h*HDP_V + d0+1)*SEQ + s] = tobf(vb);
  } else {
    Q [qrow + d0]   = (bf16_t)0.0f;
    Q [qrow + d0+1] = (bf16_t)0.0f;
    Kb[qrow + d0]   = (bf16_t)0.0f;
    Kb[qrow + d0+1] = (bf16_t)0.0f;
    if(d0 < HDP_V){
      VT[((size_t)h*HDP_V + d0  )*SEQ + s] = (bf16_t)0.0f;
      VT[((size_t)h*HDP_V + d0+1)*SEQ + s] = (bf16_t)0.0f;
    }
  }
}

// ---- attention: block = 1 head x 128 query rows, 8 waves x 16 rows each ----
__global__ void attn_kernel(const bf16_t* __restrict__ Q, const bf16_t* __restrict__ Kb,
                            const bf16_t* __restrict__ VT, const int* __restrict__ offs,
                            bf16_t* __restrict__ O){
  __shared__ alignas(16) bf16_t sQ [128*HDP_Q];     // 24576 B
  __shared__ alignas(16) bf16_t sK [2][32*HDP_Q];   // 2 x 6144 B
  __shared__ alignas(16) bf16_t sVT[2][HDP_V*32];   // 2 x 5120 B
  __shared__ alignas(16) bf16_t sP [8*16*32];       // 8192 B
  int tid=threadIdx.x, lane=tid&31, wave=tid>>5;
  int nqc = SEQ/128;
  int h = blockIdx.x / nqc;
  int qbase = (blockIdx.x % nqc)*128;
  constexpr int NT = SEQ/32;

  int off[SEGS+1];
#pragma unroll
  for(int i=0;i<SEGS+1;++i) off[i]=offs[i];

  // Q block: 1536 vec8 chunks, async
#pragma unroll
  for(int t=0;t<6;++t){
    int chunk = tid + t*256;
    async_cp16(&sQ[chunk*8], &Q[((size_t)h*SEQ + qbase)*HDP_Q + chunk*8]);
  }

  auto stageK = [&](int buf, int kt){
    int k0 = kt*32;
#pragma unroll
    for(int t=0;t<2;++t){
      int chunk = tid + t*256;
      if(chunk < 384)
        async_cp16(&sK[buf][chunk*8], &Kb[((size_t)h*SEQ + k0)*HDP_Q + chunk*8]);
    }
  };
  auto stageV = [&](int buf, int kt){
    int k0 = kt*32;
#pragma unroll
    for(int t=0;t<2;++t){
      int chunk = tid + t*256;
      if(chunk < 320){
        int r = chunk >> 2, c8 = chunk & 3;
        async_cp16(&sVT[buf][r*32 + c8*8], &VT[((size_t)h*HDP_V + r)*SEQ + k0 + c8*8]);
      }
    }
  };

  const float rscale = 0.117851130f;  // 1/sqrt(72)
  int mh = (lane>>4)*8;
  float m_i[8], s_i[8];
  int qseg[8];
#pragma unroll
  for(int i=0;i<8;++i){
    int qrow = qbase + wave*16 + mh + i;
    int sg=0;
#pragma unroll
    for(int jj=1;jj<SEGS+1;++jj) sg += (qrow >= off[jj]);
    qseg[i]=sg; m_i[i] = -1e30f; s_i[i]=0.f;
  }
  const bf16_t* aQ = sQ + wave*16*HDP_Q;

  // ---- pass 1: online row max & sum (double-buffered K tiles) ----
  stageK(0, 0); async_wait0(); __syncthreads();
  for(int kt=0; kt<NT; ++kt){
    int buf = kt & 1;
    int k0 = kt*32;
    if(kt+1 < NT) stageK(buf^1, kt+1);
#pragma unroll
    for(int half=0; half<2; ++half){
      v8f c = {};
#pragma unroll
      for(int kc=0;kc<3;++kc){
        v16bf a = load_a_frag (aQ + kc*32, HDP_Q, lane);
        v16bf b = load_bt_frag(sK[buf] + (half*16)*HDP_Q + kc*32, HDP_Q, lane);
        c = wmma_bf16(a,b,c);
      }
      int kcol = k0 + half*16 + (lane&15);
      int ksg=0;
#pragma unroll
      for(int jj=1;jj<SEGS+1;++jj) ksg += (kcol >= off[jj]);
#pragma unroll
      for(int i=0;i<8;++i){
        float val = c[i]*rscale + ((ksg==qseg[i])?1.0f:0.0f);
        float nm = fmaxf(m_i[i], val);
        s_i[i] = s_i[i]*__expf(m_i[i]-nm) + __expf(val-nm);
        m_i[i] = nm;
      }
    }
    async_wait0(); __syncthreads();
  }
  // butterfly across the 16 lanes holding each row
#pragma unroll
  for(int i=0;i<8;++i){
#pragma unroll
    for(int msk=1; msk<16; msk<<=1){
      float om = __shfl_xor(m_i[i], msk, 32);
      float os = __shfl_xor(s_i[i], msk, 32);
      float nm = fmaxf(m_i[i], om);
      s_i[i] = s_i[i]*__expf(m_i[i]-nm) + os*__expf(om-nm);
      m_i[i] = nm;
    }
  }
  float linv[8];
#pragma unroll
  for(int i=0;i<8;++i) linv[i] = 1.0f / s_i[i];

  // ---- pass 2: recompute scores, exp, P*V (double-buffered K+V tiles) ----
  v8f acc[5] = {};
  bf16_t* myP = sP + wave*16*32;
  stageK(0, 0); stageV(0, 0); async_wait0(); __syncthreads();
  for(int kt=0; kt<NT; ++kt){
    int buf = kt & 1;
    int k0 = kt*32;
    if(kt+1 < NT){ stageK(buf^1, kt+1); stageV(buf^1, kt+1); }
#pragma unroll
    for(int half=0; half<2; ++half){
      v8f c = {};
#pragma unroll
      for(int kc=0;kc<3;++kc){
        v16bf a = load_a_frag (aQ + kc*32, HDP_Q, lane);
        v16bf b = load_bt_frag(sK[buf] + (half*16)*HDP_Q + kc*32, HDP_Q, lane);
        c = wmma_bf16(a,b,c);
      }
      int kcol = k0 + half*16 + (lane&15);
      int ksg=0;
#pragma unroll
      for(int jj=1;jj<SEGS+1;++jj) ksg += (kcol >= off[jj]);
#pragma unroll
      for(int i=0;i<8;++i){
        float val = c[i]*rscale + ((ksg==qseg[i])?1.0f:0.0f);
        myP[(mh+i)*32 + half*16 + (lane&15)] = tobf(__expf(val - m_i[i]));
      }
    }
    // P is written & read by this wave only: LDS ops of a wave are in-order,
    // the compiler's dscnt wait covers the RAW; no barrier needed here.
    v16bf a = load_a_frag(myP, 32, lane);
#pragma unroll
    for(int t=0;t<5;++t){
      v16bf b = load_bt_frag(sVT[buf] + (t*16)*32, 32, lane);
      acc[t] = wmma_bf16(a, b, acc[t]);
    }
    async_wait0(); __syncthreads();
  }
#pragma unroll
  for(int t=0;t<5;++t){
    int col = t*16 + (lane&15);
    if(col < HD){
#pragma unroll
      for(int i=0;i<8;++i){
        int row = qbase + wave*16 + mh + i;
        O[(size_t)row*HID + h*HD + col] = tobf(acc[t][i]*linv[i]);
      }
    }
  }
}

extern "C" void kernel_launch(void* const* d_in, const int* in_sizes, int n_in,
                              void* d_out, int out_size, void* d_ws, size_t ws_size,
                              hipStream_t stream){
  const float* x    = (const float*)d_in[0];
  const int*   offs = (const int*)  d_in[1];
  const float* fc   = (const float*)d_in[2];
  const float* ln0g = (const float*)d_in[3];
  const float* ln0b = (const float*)d_in[4];
  const float* wqkv = (const float*)d_in[5];
  const float* bqkv = (const float*)d_in[6];
  const float* wo   = (const float*)d_in[7];
  const float* bo   = (const float*)d_in[8];
  const float* ln1g = (const float*)d_in[9];
  const float* ln1b = (const float*)d_in[10];
  const float* w1   = (const float*)d_in[11];
  const float* b1   = (const float*)d_in[12];
  const float* w2   = (const float*)d_in[13];
  const float* b2   = (const float*)d_in[14];
  float* out = (float*)d_out;

  char* ws = (char*)d_ws; size_t o = 0;
  auto alloc = [&](size_t bytes)->char*{ char* p = ws + o; o += (bytes + 255) & ~(size_t)255; return p; };
  bf16_t* wqkv_bt = (bf16_t*)alloc((size_t)HID*3*HID*2);   // [3*HID][HID]
  bf16_t* wo_bt   = (bf16_t*)alloc((size_t)HID*HID*2);     // [HID][HID]
  bf16_t* w1_bt   = (bf16_t*)alloc((size_t)HID*MLP_D*2);   // [MLP_D][HID]
  bf16_t* w2_bt   = (bf16_t*)alloc((size_t)MLP_D*HID*2);   // [HID][MLP_D]
  bf16_t* ln0_bf  = (bf16_t*)alloc((size_t)SEQ*HID*2);
  float*  qkv     = (float*) alloc((size_t)SEQ*3*HID*4);
  bf16_t* Qb      = (bf16_t*)alloc((size_t)NH*SEQ*HDP_Q*2);
  bf16_t* Kbb     = (bf16_t*)alloc((size_t)NH*SEQ*HDP_Q*2);
  bf16_t* VTb     = (bf16_t*)alloc((size_t)NH*HDP_V*SEQ*2);
  bf16_t* attn_bf = (bf16_t*)alloc((size_t)SEQ*HID*2);
  float*  x1      = (float*) alloc((size_t)SEQ*HID*4);
  bf16_t* ln1_bf  = (bf16_t*)alloc((size_t)SEQ*HID*2);
  bf16_t* h_bf    = (bf16_t*)alloc((size_t)SEQ*MLP_D*2);
  (void)ws_size; (void)in_sizes; (void)n_in; (void)out_size;

  // weight convert + transpose to [N][K] bf16 (once per launch; weights static)
  { dim3 g((HID+31)/32, (3*HID+31)/32);
    cvt_transpose_bf16_kernel<<<g,256,0,stream>>>(wqkv, wqkv_bt, HID, 3*HID); }
  { dim3 g((HID+31)/32, (HID+31)/32);
    cvt_transpose_bf16_kernel<<<g,256,0,stream>>>(wo, wo_bt, HID, HID); }
  { dim3 g((HID+31)/32, (MLP_D+31)/32);
    cvt_transpose_bf16_kernel<<<g,256,0,stream>>>(w1, w1_bt, HID, MLP_D); }
  { dim3 g((MLP_D+31)/32, (HID+31)/32);
    cvt_transpose_bf16_kernel<<<g,256,0,stream>>>(w2, w2_bt, MLP_D, HID); }

  ln_bf16_kernel<<<SEQ,256,0,stream>>>(x, ln0g, ln0b, ln0_bf);

  { dim3 g((3*HID+127)/128, SEQ/64);
    gemm_bf16_kernel<0><<<g,256,0,stream>>>(ln0_bf, wqkv_bt, bqkv, nullptr, qkv, nullptr, SEQ, 3*HID, HID); }

  { int total = SEQ*NH*48;
    rope_scatter_kernel<<<(total+255)/256,256,0,stream>>>(qkv, fc, Qb, Kbb, VTb); }

  attn_kernel<<<NH*(SEQ/128),256,0,stream>>>(Qb, Kbb, VTb, offs, attn_bf);

  { dim3 g((HID+127)/128, SEQ/64);
    gemm_bf16_kernel<1><<<g,256,0,stream>>>(attn_bf, wo_bt, bo, x, x1, nullptr, SEQ, HID, HID); }

  ln_bf16_kernel<<<SEQ,256,0,stream>>>(x1, ln1g, ln1b, ln1_bf);

  { dim3 g((MLP_D+127)/128, SEQ/64);
    gemm_bf16_kernel<2><<<g,256,0,stream>>>(ln1_bf, w1_bt, b1, nullptr, nullptr, h_bf, SEQ, MLP_D, HID); }

  { dim3 g((HID+127)/128, SEQ/64);
    gemm_bf16_kernel<1><<<g,256,0,stream>>>(h_bf, w2_bt, b2, x1, out, nullptr, SEQ, HID, MLP_D); }
}
